// Retriever_23811298689257
// MI455X (gfx1250) — compile-verified, matrix operand order checked by reading, not run on hardware
//
#include <hip/hip_runtime.h>
#include <hip/hip_bf16.h>

// ---------------------------------------------------------------------------
// Types for CDNA5 WMMA
// ---------------------------------------------------------------------------
typedef __attribute__((ext_vector_type(16))) __bf16 v16bf;
typedef __attribute__((ext_vector_type(8)))  float  v8f;

union BfFrag {
    int4  q[2];   // 2 x 16B loads = 32B = 16 bf16
    v16bf v;
};

#define EMBD   128
#define LDK    160   // 138 padded to multiple of 32
#define KTILES 5     // 160 / 32

// ---------------------------------------------------------------------------
// f32 -> bf16 (round to nearest even, ignores NaN corner)
// ---------------------------------------------------------------------------
__device__ __forceinline__ unsigned short f2bf(float x) {
    unsigned int u = __float_as_uint(x);
    u += 0x7fffu + ((u >> 16) & 1u);
    return (unsigned short)(u >> 16);
}

// ---------------------------------------------------------------------------
// generic float zero
// ---------------------------------------------------------------------------
__global__ void k_zero_f32(float* p, long n) {
    long i = (long)blockIdx.x * blockDim.x + threadIdx.x;
    long stride = (long)gridDim.x * blockDim.x;
    for (; i < n; i += stride) p[i] = 0.0f;
}

// ---------------------------------------------------------------------------
// degree counts (by dst of each direction)
// ---------------------------------------------------------------------------
__global__ void k_count_deg(const int* __restrict__ h, const int* __restrict__ t,
                            float* deg_h, float* deg_t, int E) {
    int e = blockIdx.x * blockDim.x + threadIdx.x;
    if (e >= E) return;
    atomicAdd(&deg_t[t[e]], 1.0f);
    atomicAdd(&deg_h[h[e]], 1.0f);
}

// ---------------------------------------------------------------------------
// scatter-add of [N,2] features along edges (out must be pre-zeroed)
// ---------------------------------------------------------------------------
__global__ void k_scatter_add2(const float* __restrict__ xin,
                               const int* __restrict__ src,
                               const int* __restrict__ dst,
                               float* __restrict__ xout, int E) {
    int e = blockIdx.x * blockDim.x + threadIdx.x;
    if (e >= E) return;
    int s = src[e], d = dst[e];
    float a = xin[2 * s];
    float b = xin[2 * s + 1];
    atomicAdd(&xout[2 * d],     a);
    atomicAdd(&xout[2 * d + 1], b);
}

__global__ void k_div_deg2(float* __restrict__ x, const float* __restrict__ deg, int N) {
    int n = blockIdx.x * blockDim.x + threadIdx.x;
    if (n >= N) return;
    float inv = 1.0f / fmaxf(deg[n], 1.0f);
    x[2 * n]     *= inv;
    x[2 * n + 1] *= inv;
}

// ---------------------------------------------------------------------------
// Convert W1 h-block / t-block into bf16 [128 x LDK] row-major (K padded)
// W1 is [128 x 532]: q: cols 0..127, h: 128..265, r: 266..393, t: 394..531
// ---------------------------------------------------------------------------
__global__ void k_conv_w(const float* __restrict__ W1,
                         unsigned short* __restrict__ Wh,
                         unsigned short* __restrict__ Wt) {
    int idx = blockIdx.x * blockDim.x + threadIdx.x;
    if (idx >= 128 * LDK) return;
    int n = idx / LDK;
    int k = idx % LDK;
    float vh = (k < 138) ? W1[n * 532 + 128 + k] : 0.0f;
    float vt = (k < 138) ? W1[n * 532 + 394 + k] : 0.0f;
    Wh[idx] = f2bf(vh);
    Wt[idx] = f2bf(vt);
}

// cqb[j] = b1[j] + sum_k W1[j, k] * q[k]   (q-block folded into bias)
__global__ void k_cqb(const float* __restrict__ W1, const float* __restrict__ b1,
                      const float* __restrict__ q, float* __restrict__ cqb) {
    int j = blockIdx.x * blockDim.x + threadIdx.x;
    if (j >= 128) return;
    float s = b1[j];
    const float* row = W1 + (size_t)j * 532;
    #pragma unroll 4
    for (int k = 0; k < 128; ++k) s += row[k] * q[k];
    cqb[j] = s;
}

// rel_proj[r, j] = sum_k W1[j, 266+k] * rel[r, k]
__global__ void k_relproj(const float* __restrict__ W1, const float* __restrict__ rel,
                          float* __restrict__ rp, int R) {
    int idx = blockIdx.x * blockDim.x + threadIdx.x;
    if (idx >= R * 128) return;
    int r = idx / 128;
    int j = idx % 128;
    const float* wrow = W1 + (size_t)j * 532 + 266;
    const float* rrow = rel + (size_t)r * 128;
    float s = 0.0f;
    #pragma unroll 4
    for (int k = 0; k < 128; ++k) s += wrow[k] * rrow[k];
    rp[(size_t)r * 128 + j] = s;
}

// ---------------------------------------------------------------------------
// Build h_full bf16 [Npad x LDK]:
// cols 0..127: entity or non-text row; 128..129 topic; 130..137 dde1..4; rest 0
// ---------------------------------------------------------------------------
__global__ void k_build_hfull(const float* __restrict__ ent, const float* __restrict__ ntx,
                              const float* __restrict__ topic,
                              const float* __restrict__ d1, const float* __restrict__ d2,
                              const float* __restrict__ d3, const float* __restrict__ d4,
                              unsigned short* __restrict__ hf,
                              int N, int NT, long total) {
    long idx = (long)blockIdx.x * blockDim.x + threadIdx.x;
    if (idx >= total) return;
    int n = (int)(idx / LDK);
    int c = (int)(idx % LDK);
    float v = 0.0f;
    if (n < N) {
        if (c < 128)       v = (n < NT) ? ent[(size_t)n * 128 + c] : ntx[c];
        else if (c < 130)  v = topic[2 * n + (c - 128)];
        else if (c < 132)  v = d1[2 * n + (c - 130)];
        else if (c < 134)  v = d2[2 * n + (c - 132)];
        else if (c < 136)  v = d3[2 * n + (c - 134)];
        else if (c < 138)  v = d4[2 * n + (c - 136)];
    }
    hf[idx] = f2bf(v);
}

// ---------------------------------------------------------------------------
// WMMA GEMM: C[Npad x 128] = A[Npad x LDK](bf16) @ B[128 x LDK](bf16)^T
// One 16x16 output tile per wave; 8 waves/block cover all 128 columns.
// A rows AND C rows are padded to a multiple of 16, so no guards anywhere:
// pure load -> wmma -> store.
// Fragment addressing per CDNA5 ISA 16-bit operand layout:
//   lanes 0-15 : K 0-7 (v0..v3) and 16-23 (v4..v7)
//   lanes 16-31: K 8-15         and 24-31
// ---------------------------------------------------------------------------
__global__ __launch_bounds__(256)
void k_gemm_proj(const unsigned short* __restrict__ A,
                 const unsigned short* __restrict__ Bm,
                 float* __restrict__ C) {
    const int mBase = blockIdx.x * 16;
    const int wave  = threadIdx.x >> 5;
    const int lane  = threadIdx.x & 31;
    const int nBase = wave * 16;

    const int rA  = mBase + (lane & 15);
    const int rB  = nBase + (lane & 15);
    const int hiK = (lane >> 4) * 8;

    v8f acc = {};
    #pragma unroll
    for (int kk = 0; kk < KTILES; ++kk) {
        const int kb = kk * 32 + hiK;
        BfFrag a, b;
        a.q[0] = *(const int4*)(A  + (size_t)rA * LDK + kb);
        a.q[1] = *(const int4*)(A  + (size_t)rA * LDK + kb + 16);
        b.q[0] = *(const int4*)(Bm + (size_t)rB * LDK + kb);
        b.q[1] = *(const int4*)(Bm + (size_t)rB * LDK + kb + 16);
        acc = __builtin_amdgcn_wmma_f32_16x16x32_bf16(
            /*neg_a=*/false, a.v, /*neg_b=*/false, b.v,
            /*c_mod=*/(short)0, acc, /*reuse_a=*/false, /*reuse_b=*/false);
    }

    // D layout: vgpr v holds M = v (lanes 0-15) / v+8 (lanes 16-31), N = lane&15
    const int rowHi = (lane >> 4) * 8;
    const int col   = nBase + (lane & 15);
    float* cbase = C + (size_t)(mBase + rowHi) * 128 + col;
    #pragma unroll
    for (int v = 0; v < 8; ++v) {
        cbase[(size_t)v * 128] = acc[v];
    }
}

// ---------------------------------------------------------------------------
// Edge pass: one wave per edge.
// pred[e] = b2 + sum_j W2[j]*relu(cqb[j] + ph[h][j] + pr[r][j] + pt[t][j])
// lane handles 4 dims via float4 -> 512B coalesced per table row (L2 resident).
// ---------------------------------------------------------------------------
__global__ __launch_bounds__(256)
void k_edge_pred(const int* __restrict__ h, const int* __restrict__ r,
                 const int* __restrict__ t,
                 const float* __restrict__ ph, const float* __restrict__ pt,
                 const float* __restrict__ pr, const float* __restrict__ cqb,
                 const float* __restrict__ W2, const float* __restrict__ b2,
                 float* __restrict__ out, int E) {
    int wid  = (blockIdx.x * blockDim.x + threadIdx.x) >> 5;
    int lane = threadIdx.x & 31;
    if (wid >= E) return;

    int hh = h[wid], rr = r[wid], tt = t[wid];

    float4 a = *(const float4*)(ph  + (size_t)hh * 128 + lane * 4);
    float4 b = *(const float4*)(pt  + (size_t)tt * 128 + lane * 4);
    float4 c = *(const float4*)(pr  + (size_t)rr * 128 + lane * 4);
    float4 q = ((const float4*)cqb)[lane];
    float4 w = ((const float4*)W2)[lane];

    float s = fmaxf(q.x + a.x + b.x + c.x, 0.0f) * w.x
            + fmaxf(q.y + a.y + b.y + c.y, 0.0f) * w.y
            + fmaxf(q.z + a.z + b.z + c.z, 0.0f) * w.z
            + fmaxf(q.w + a.w + b.w + c.w, 0.0f) * w.w;

    #pragma unroll
    for (int off = 16; off > 0; off >>= 1)
        s += __shfl_xor(s, off, 32);

    if (lane == 0) out[wid] = s + b2[0];
}

// ---------------------------------------------------------------------------
// Host launcher
// ---------------------------------------------------------------------------
extern "C" void kernel_launch(void* const* d_in, const int* in_sizes, int n_in,
                              void* d_out, int out_size, void* d_ws, size_t ws_size,
                              hipStream_t stream) {
    const int*   h_id   = (const int*)  d_in[0];
    const int*   r_id   = (const int*)  d_in[1];
    const int*   t_id   = (const int*)  d_in[2];
    const float* q_emb  = (const float*)d_in[3];
    const float* ent    = (const float*)d_in[4];
    const float* rel    = (const float*)d_in[6];
    const float* topic  = (const float*)d_in[7];
    const float* ntx    = (const float*)d_in[8];
    const float* W1     = (const float*)d_in[9];
    const float* b1     = (const float*)d_in[10];
    const float* W2     = (const float*)d_in[11];
    const float* b2     = (const float*)d_in[12];
    float*       out    = (float*)d_out;

    const int E  = in_sizes[0];
    const int NT = in_sizes[4] / EMBD;          // text entities
    const int R  = in_sizes[6] / EMBD;          // relations
    const int N  = in_sizes[7] / 2;             // total nodes (topic one-hot rows)
    const int Mtiles = (N + 15) / 16;
    const int Npad   = Mtiles * 16;

    // ---- workspace carve-up (256B aligned) ----
    size_t o = 0;
    auto take = [&](size_t bytes) { size_t p = o; o += (bytes + 255) & ~(size_t)255; return p; };
    char* ws = (char*)d_ws;

    float*          proj_h = (float*)(ws + take((size_t)Npad * 128 * 4));
    float*          proj_t = (float*)(ws + take((size_t)Npad * 128 * 4));
    unsigned short* hfull  = (unsigned short*)(ws + take((size_t)Npad * LDK * 2));
    unsigned short* Wh     = (unsigned short*)(ws + take((size_t)128 * LDK * 2));
    unsigned short* Wt     = (unsigned short*)(ws + take((size_t)128 * LDK * 2));
    float*          rproj  = (float*)(ws + take((size_t)R * 128 * 4));
    float*          cqb    = (float*)(ws + take(128 * 4));
    // contiguous zero region: deg_h, deg_t, dde1..4  -> 10*N floats
    float*          zreg   = (float*)(ws + take((size_t)10 * N * 4));
    float* deg_h = zreg;
    float* deg_t = zreg + (size_t)N;
    float* dde1  = zreg + (size_t)2 * N;
    float* dde2  = zreg + (size_t)4 * N;
    float* dde3  = zreg + (size_t)6 * N;
    float* dde4  = zreg + (size_t)8 * N;

    const int T = 256;
    auto cdiv = [](long a, long b) { return (int)((a + b - 1) / b); };

    // 1) zero degrees + dde accumulators
    k_zero_f32<<<cdiv((long)10 * N, T), T, 0, stream>>>(zreg, (long)10 * N);

    // 2) degree counts
    k_count_deg<<<cdiv(E, T), T, 0, stream>>>(h_id, t_id, deg_h, deg_t, E);

    // 3) DDE rounds: fwd (h->t) x2, rev (t->h) x2
    k_scatter_add2<<<cdiv(E, T), T, 0, stream>>>(topic, h_id, t_id, dde1, E);
    k_div_deg2    <<<cdiv(N, T), T, 0, stream>>>(dde1, deg_t, N);
    k_scatter_add2<<<cdiv(E, T), T, 0, stream>>>(dde1,  h_id, t_id, dde2, E);
    k_div_deg2    <<<cdiv(N, T), T, 0, stream>>>(dde2, deg_t, N);
    k_scatter_add2<<<cdiv(E, T), T, 0, stream>>>(topic, t_id, h_id, dde3, E);
    k_div_deg2    <<<cdiv(N, T), T, 0, stream>>>(dde3, deg_h, N);
    k_scatter_add2<<<cdiv(E, T), T, 0, stream>>>(dde3,  t_id, h_id, dde4, E);
    k_div_deg2    <<<cdiv(N, T), T, 0, stream>>>(dde4, deg_h, N);

    // 4) weight prep: bf16 W-blocks, folded q-bias, relation projections
    k_conv_w <<<cdiv(128 * LDK, T), T, 0, stream>>>(W1, Wh, Wt);
    k_cqb    <<<1, 128, 0, stream>>>(W1, b1, q_emb, cqb);
    k_relproj<<<cdiv((long)R * 128, T), T, 0, stream>>>(W1, rel, rproj, R);

    // 5) build bf16 node-feature table (padded rows zeroed)
    long total = (long)Npad * LDK;
    k_build_hfull<<<cdiv(total, T), T, 0, stream>>>(ent, ntx, topic,
                                                    dde1, dde2, dde3, dde4,
                                                    hfull, N, NT, total);

    // 6) WMMA node projections: proj_h = hfull @ Wh^T, proj_t = hfull @ Wt^T
    k_gemm_proj<<<Mtiles, 256, 0, stream>>>(hfull, Wh, proj_h);
    k_gemm_proj<<<Mtiles, 256, 0, stream>>>(hfull, Wt, proj_t);

    // 7) edge pass: gather + relu + dot(W2), one wave per edge
    k_edge_pred<<<cdiv((long)E * 32, T), T, 0, stream>>>(h_id, r_id, t_id,
                                                         proj_h, proj_t, rproj,
                                                         cqb, W2, b2, out, E);
}
